// LSTMReloadNet_52123723104278
// MI455X (gfx1250) — matrix operand order repaired and checked by reading, not compiled, not run on hardware
//
#include <hip/hip_runtime.h>
#include <hip/hip_bf16.h>

// ---------------------------------------------------------------------------
// Fused LSTM (B=4096, T=256, F=64, H=32) + FC head for MI455X (gfx1250).
//
// Design (compile-only reasoning, see analysis):
//  * One wave32 per 16-row batch tile -> 256 blocks x 32 threads.
//  * W_ih / W_hh live in registers as bf16 WMMA B-fragments for the whole
//    T=256 loop (16 + 8 fragments). Accum in f32 via
//    v_wmma_f32_16x16x32_bf16 (24 WMMA per timestep per tile).
//  * x_t is streamed straight from HBM (268 MB total = the roofline floor),
//    converted to bf16 A-fragments in-register; global_prefetch_b8 for t+1.
//  * Gate elementwise is pure per-lane register math (i/f/g/o for a given
//    (m,hc) share lane+slot across acc fragments). c stays in VGPRs.
//  * h does an in-wave LDS round trip for the C-layout -> A-layout transpose.
// ---------------------------------------------------------------------------

#define Bn 4096
#define Tn 256
#define Fn 64
#define Hn 32
#define Gn 128   // 4*H

typedef __attribute__((ext_vector_type(16))) __bf16 v16bf;
typedef __attribute__((ext_vector_type(8)))  float  v8f;

__device__ __forceinline__ float fast_sigmoid(float x) {
    return __builtin_amdgcn_rcpf(1.0f + __expf(-x));
}
__device__ __forceinline__ float fast_tanh(float x) {
    // tanh(x) = 1 - 2/(exp(2x)+1)
    return 1.0f - 2.0f * __builtin_amdgcn_rcpf(1.0f + __expf(2.0f * x));
}

// A-fragment K offset for VGPR-pair slot i (16-bit A matrix 16x32, ISA 7.12.2)
__device__ __forceinline__ int a_koff(int i) { return (i < 4) ? 2 * i : 2 * i + 8; }

__global__ __launch_bounds__(32)
void lstm_fused_wmma(const float* __restrict__ x,
                     const float* __restrict__ Wih,
                     const float* __restrict__ Whh,
                     const float* __restrict__ bih,
                     const float* __restrict__ bhh,
                     const float* __restrict__ W1,
                     const float* __restrict__ b1,
                     const float* __restrict__ W2,
                     const float* __restrict__ b2,
                     float* __restrict__ out)
{
    const int lane = threadIdx.x;      // 0..31 (wave32)
    const int l16  = lane & 15;
    const int lh   = lane >> 4;        // which half of the wave
    const int b0   = blockIdx.x * 16;  // batch tile start

    // h state, padded row stride 34 floats to dodge LDS bank conflicts
    __shared__ float hls[16 * 34];
    for (int i = lane; i < 16 * 34; i += 32) hls[i] = 0.0f;

    // ---- Load weight B-fragments (bf16) + fused bias, once ----------------
    // B-matrix 32x16 layout: lane -> N=l16; VGPR slot i holds K pair
    // {a_koff(i),+1} + 8*lh (+32*kt for the K=64 input projection).
    v16bf wihf[2][8];   // [k-tile][n-tile]
    v16bf whhf[8];      // [n-tile]
    float bias[8];
    #pragma unroll
    for (int n = 0; n < 8; ++n) {
        const int gcol = n * 16 + l16;          // gate column 0..127
        bias[n] = bih[gcol] + bhh[gcol];
        #pragma unroll
        for (int i = 0; i < 8; ++i) {
            const int kb = 8 * lh + a_koff(i);
            float2 wh = *(const float2*)(Whh + gcol * Hn + kb);
            whhf[n][2 * i]     = (__bf16)wh.x;
            whhf[n][2 * i + 1] = (__bf16)wh.y;
            #pragma unroll
            for (int kt = 0; kt < 2; ++kt) {
                float2 wx = *(const float2*)(Wih + gcol * Fn + 32 * kt + kb);
                wihf[kt][n][2 * i]     = (__bf16)wx.x;
                wihf[kt][n][2 * i + 1] = (__bf16)wx.y;
            }
        }
    }

    // cell state in registers: cst[j][r] <-> (m = r+8*lh, hc = 16j + l16)
    float cst[2][8];
    #pragma unroll
    for (int j = 0; j < 2; ++j)
        #pragma unroll
        for (int r = 0; r < 8; ++r) cst[j][r] = 0.0f;

    __syncthreads();

    // per-lane row base into x: this lane owns batch row (b0 + l16)
    const float* pxl = x + (size_t)(b0 + l16) * Tn * Fn;

    for (int t = 0; t < Tn; ++t) {
        // Prefetch next timestep's x rows (global_prefetch_b8).
        if (t + 1 < Tn)
            __builtin_prefetch(pxl + (size_t)(t + 1) * Fn + lh * 32, 0, 0);

        // ---- Build bf16 A-fragments for x_t (16x64 -> two 16x32 tiles) ----
        // A layout: lane -> M=l16; slot i holds K pair {a_koff(i), +1} + 8*lh.
        v16bf ax[2];
        #pragma unroll
        for (int kt = 0; kt < 2; ++kt) {
            #pragma unroll
            for (int i = 0; i < 8; ++i) {
                const int k = 32 * kt + 8 * lh + a_koff(i);
                float2 v = *(const float2*)(pxl + (size_t)t * Fn + k);
                ax[kt][2 * i]     = (__bf16)v.x;
                ax[kt][2 * i + 1] = (__bf16)v.y;
            }
        }
        // ---- Build bf16 A-fragment for h_{t-1} (16x32) from LDS -----------
        v16bf ah;
        #pragma unroll
        for (int i = 0; i < 8; ++i) {
            const int k = 8 * lh + a_koff(i);
            float2 v = *(const float2*)(hls + l16 * 34 + k);
            ah[2 * i]     = (__bf16)v.x;
            ah[2 * i + 1] = (__bf16)v.y;
        }

        // ---- gates = bias + x_t@W_ih^T + h@W_hh^T : 24 WMMAs --------------
        v8f acc[8];
        #pragma unroll
        for (int n = 0; n < 8; ++n) {
            v8f a;
            #pragma unroll
            for (int r = 0; r < 8; ++r) a[r] = bias[n];
            a = __builtin_amdgcn_wmma_f32_16x16x32_bf16(
                    false, ax[0], false, wihf[0][n], (short)0, a, false, false);
            a = __builtin_amdgcn_wmma_f32_16x16x32_bf16(
                    false, ax[1], false, wihf[1][n], (short)0, a, false, false);
            a = __builtin_amdgcn_wmma_f32_16x16x32_bf16(
                    false, ah,    false, whhf[n],    (short)0, a, false, false);
            acc[n] = a;
        }

        // ---- LSTM pointwise: pure per-lane register math ------------------
        // Element (m = r+8*lh, hc = 16j+l16): i=acc[j], f=acc[2+j],
        // g=acc[4+j], o=acc[6+j], all at slot r of this lane.
        #pragma unroll
        for (int j = 0; j < 2; ++j) {
            #pragma unroll
            for (int r = 0; r < 8; ++r) {
                const float ig = fast_sigmoid(acc[0 + j][r]);
                const float fg = fast_sigmoid(acc[2 + j][r]);
                const float gg = fast_tanh   (acc[4 + j][r]);
                const float og = fast_sigmoid(acc[6 + j][r]);
                const float c  = fg * cst[j][r] + ig * gg;
                cst[j][r] = c;
                const float h = og * fast_tanh(c);
                hls[(r + 8 * lh) * 34 + 16 * j + l16] = h;  // C-layout -> LDS
            }
        }
        __syncthreads();  // single-wave WG: orders LDS (h) for next step
    }

    // ---- FC head: relu -> (32->16) -> relu -> (16->1) -> sigmoid ----------
    __syncthreads();
    if (lane < 16) {
        const int m = lane;
        float zr[32];
        #pragma unroll
        for (int k = 0; k < 32; ++k) {
            float h = hls[m * 34 + k];
            zr[k] = h > 0.0f ? h : 0.0f;
        }
        float o = b2[0];
        #pragma unroll
        for (int q = 0; q < 16; ++q) {
            float s = b1[q];
            #pragma unroll
            for (int k = 0; k < 32; ++k)
                s = fmaf(zr[k], W1[q * 32 + k], s);
            s = s > 0.0f ? s : 0.0f;
            o = fmaf(s, W2[q], o);
        }
        out[b0 + m] = fast_sigmoid(o);
    }
}

extern "C" void kernel_launch(void* const* d_in, const int* in_sizes, int n_in,
                              void* d_out, int out_size, void* d_ws, size_t ws_size,
                              hipStream_t stream) {
    const float* x   = (const float*)d_in[0];
    const float* Wih = (const float*)d_in[1];
    const float* Whh = (const float*)d_in[2];
    const float* bih = (const float*)d_in[3];
    const float* bhh = (const float*)d_in[4];
    const float* W1  = (const float*)d_in[5];
    const float* b1  = (const float*)d_in[6];
    const float* W2  = (const float*)d_in[7];
    const float* b2  = (const float*)d_in[8];
    float* out = (float*)d_out;

    dim3 grid(Bn / 16), block(32);
    hipLaunchKernelGGL(lstm_fused_wmma, grid, block, 0, stream,
                       x, Wih, Whh, bih, bhh, W1, b1, W2, b2, out);
}
